// MultiHeadedAttention_43344809951270
// MI455X (gfx1250) — compile-verified, matrix operand order checked by reading, not run on hardware
//
#include <hip/hip_runtime.h>
#include <hip/hip_bf16.h>

// Problem constants (match reference): B=4, T=2048, E=768, H=3, D=256
#define BB 4
#define TT 2048
#define EE 768
#define HH 3
#define DD 256

typedef __attribute__((ext_vector_type(16))) __bf16 v16bf;
typedef __attribute__((ext_vector_type(8)))  __bf16 v8bf;
typedef __attribute__((ext_vector_type(8)))  float  v8f;

#define CAT16(lo, hi) \
  __builtin_shufflevector(lo, hi, 0, 1, 2, 3, 4, 5, 6, 7, 8, 9, 10, 11, 12, 13, 14, 15)

__device__ __forceinline__ v8f wmma_bf16(const v16bf a, const v16bf b, const v8f c) {
  return __builtin_amdgcn_wmma_f32_16x16x32_bf16(
      /*neg_a=*/false, a, /*neg_b=*/false, b,
      /*c_mod=*/(short)0, c, /*reuse_a=*/false, /*reuse_b=*/false);
}

// Load one 16-lane-striped bf16 A/B fragment: two contiguous 16B chunks at
// ptr[kb .. kb+7] and ptr[kb+16 .. kb+23] (kb = 0 for lanes 0-15, 8 for 16-31).
__device__ __forceinline__ v16bf load_frag(const __bf16* p, int kb) {
  const v8bf lo = *(const v8bf*)(p + kb);
  const v8bf hi = *(const v8bf*)(p + kb + 16);
  return CAT16(lo, hi);
}

// ---- CDNA5 async global->LDS staging (ASYNCcnt path), with safe fallback ---
#if defined(__has_builtin)
#if __has_builtin(__builtin_amdgcn_global_load_async_to_lds_b128)
#define ASYNC_LDS 1
#endif
#endif

// GCC-style int4 (matches the builtin's V4i canonical type exactly)
typedef int v4i __attribute__((vector_size(16)));
typedef v4i __attribute__((address_space(1)))* as1_v4ip;   // global int4*
typedef v4i __attribute__((address_space(3)))* as3_v4ip;   // LDS int4*

__device__ __forceinline__ void stage16(__bf16* ldst, const __bf16* gsrc) {
#ifdef ASYNC_LDS
  __builtin_amdgcn_global_load_async_to_lds_b128(
      (as1_v4ip)(unsigned long long)gsrc,
      (as3_v4ip)(unsigned)(unsigned long long)ldst, 0, 0);
#else
  *(v8bf*)ldst = *(const v8bf*)gsrc;
#endif
}

// ---------------------------------------------------------------------------
// Kernel 1: per-head QKV projection.
// Block = 8 waves sharing one 64-wide weight column tile; per k-step the block
// stages W[32k x 64d] into LDS transposed to [d][k] bf16, then each wave does
// 4 WMMAs producing a 16x64 strip.  Q,K -> [B][H][T][D]; V -> [B][H][D][T].
// ---------------------------------------------------------------------------
__global__ __launch_bounds__(256) void qkv_proj_kernel(
    const float* __restrict__ Xk, const float* __restrict__ Xv,
    const float* __restrict__ Xq,
    const float* __restrict__ Wk, const float* __restrict__ Wq,
    const float* __restrict__ Wv,
    __bf16* __restrict__ Qb, __bf16* __restrict__ Kb, __bf16* __restrict__ Vt)
{
  __shared__ __align__(16) __bf16 wlds[64 * 32];   // [d within tile][k] 4KB

  const int tid  = threadIdx.x;
  const int lane = tid & 31;
  const int wave = tid >> 5;

  unsigned id = blockIdx.x;
  const int tg   = id % (TT / 16 / 8); id /= (TT / 16 / 8);   // 16 t-groups
  const int b    = id % BB;            id /= BB;
  const int d64  = id % (DD / 64);     id /= (DD / 64);       // 4 col tiles
  const int h    = id % HH;            id /= HH;
  const int type = id;                                        // 0=Q 1=K 2=V

  const float* X = (type == 0) ? Xq : (type == 1) ? Xk : Xv;
  const float* W = (type == 0) ? Wq : (type == 1) ? Wk : Wv;

  const int half = lane >> 4, ln = lane & 15, kb = half ? 8 : 0;

  const int    tt   = tg * 8 + wave;
  const int    trow = tt * 16 + ln;                 // A-matrix row
  const float* Xrow = X + ((size_t)b * TT + trow) * EE;

  // Weight staging role: thread -> (column dl of 64, k-octet kq of 4)
  const int dl = tid & 63;
  const int kq = tid >> 6;
  const float* Wcol = W + (size_t)h * EE * DD + (size_t)d64 * 64 + dl;

  v8f acc[4];
  v8f zero = {};
#pragma unroll
  for (int i = 0; i < 4; ++i) acc[i] = zero;

  for (int e0 = 0; e0 < EE; e0 += 32) {
    // cooperative stage: W[e0+kq*8 .. +7][dcol] -> wlds[dl][kq*8 ..] (bf16)
    v8bf wv;
#pragma unroll
    for (int j = 0; j < 8; ++j)
      wv[j] = (__bf16)Wcol[(size_t)(e0 + kq * 8 + j) * DD];
    *(v8bf*)(wlds + dl * 32 + kq * 8) = wv;
    __syncthreads();

    // A fragment from f32 activations (two 32B contiguous groups per lane)
    v16bf a;
#pragma unroll
    for (int j = 0; j < 8; ++j) {
      a[j]     = (__bf16)Xrow[e0 + kb + j];
      a[8 + j] = (__bf16)Xrow[e0 + kb + 16 + j];
    }
#pragma unroll
    for (int i = 0; i < 4; ++i) {
      const v16bf bm = load_frag(wlds + (i * 16 + ln) * 32, kb);
      acc[i] = wmma_bf16(a, bm, acc[i]);
    }
    __syncthreads();
  }

  const size_t bh = (size_t)b * HH + h;
  if (type != 2) {                                  // Q or K: [B][H][T][D]
    __bf16* Out = (type == 0) ? Qb : Kb;
#pragma unroll
    for (int i = 0; i < 4; ++i) {
      const int dcol = d64 * 64 + i * 16 + ln;
#pragma unroll
      for (int rr = 0; rr < 8; ++rr) {
        const int m = tt * 16 + rr + half * 8;
        Out[(bh * TT + m) * DD + dcol] = (__bf16)acc[i][rr];
      }
    }
  } else {                                          // V transposed: [B][H][D][T]
#pragma unroll
    for (int i = 0; i < 4; ++i) {
      const int dcol = d64 * 64 + i * 16 + ln;
      v8bf sv;
#pragma unroll
      for (int rr = 0; rr < 8; ++rr) sv[rr] = (__bf16)acc[i][rr];
      *(v8bf*)(Vt + (bh * DD + dcol) * TT + tt * 16 + half * 8) = sv;
    }
  }
}

// ---------------------------------------------------------------------------
// Kernel 2: causal flash attention.  4 waves (adjacent q-tiles, same (b,h))
// per block cooperatively stage each 32-key K/V panel into LDS (async-to-LDS
// when available), then every wave runs QK^T + online softmax + P*V out of
// LDS.  One wave owns 16 query rows and all of D=256 in f32 accumulators.
// ---------------------------------------------------------------------------
__global__ __launch_bounds__(128) void attn_kernel(
    const __bf16* __restrict__ Qb, const __bf16* __restrict__ Kb,
    const __bf16* __restrict__ Vt, __bf16* __restrict__ Zb)
{
  __shared__ __align__(16) __bf16 klds[32 * DD];      // K panel  [t][d]   16KB
  __shared__ __align__(16) __bf16 vlds[DD * 32];      // V^T panel [d][t]  16KB
  __shared__ __align__(16) __bf16 plds[4][16 * 32];   // per-wave P staging 4KB

  const int tid   = threadIdx.x;
  const int lane  = tid & 31;
  const int wslot = tid >> 5;
  const unsigned wid = blockIdx.x * 4u + wslot;

  const int qt = wid % (TT / 16);
  unsigned  r  = wid / (TT / 16);            // same for all 4 waves (128 % 4 == 0)
  const int h  = r % HH;
  const int b  = r / HH;

  const int half = lane >> 4, ln = lane & 15, kb = half ? 8 : 0;
  const size_t bh = (size_t)b * HH + h;

  // Preload the Q stripe (16 rows x D=256) as 8 A-fragments.
  const __bf16* Qrow = Qb + (bh * TT + qt * 16 + ln) * DD;
  v16bf qf[8];
#pragma unroll
  for (int dc = 0; dc < 8; ++dc) qf[dc] = load_frag(Qrow + dc * 32, kb);

  v8f o[16];
  v8f zero = {};
#pragma unroll
  for (int nt = 0; nt < 16; ++nt) o[nt] = zero;
  float mrow[8], lrow[8];
#pragma unroll
  for (int rr = 0; rr < 8; ++rr) { mrow[rr] = -1e30f; lrow[rr] = 0.f; }

  const float sc    = 0.02209708691207961f;   // 1/sqrt(T), reference scaling
  const float LOG2E = 1.4426950408889634f;
  const int   qt_base = (blockIdx.x * 4) % (TT / 16);
  const int   nkb     = qt / 2 + 1;               // this wave's causal blocks
  const int   nkb_max = (qt_base + 3) / 2 + 1;    // block-wide loop bound
  const int   qrow0   = qt * 16 + half * 8;

  for (int kt = 0; kt < nkb_max; ++kt) {
    // ---- cooperative stage of this 32-key K/V panel into LDS --------------
    const __bf16* Kg = Kb + (bh * TT + (size_t)kt * 32) * DD;   // 16KB contig
    const __bf16* Vg = Vt + bh * DD * TT + (size_t)kt * 32;     // 256 x 64B rows
#pragma unroll
    for (int i = 0; i < 8; ++i) {
      const int c = tid + 128 * i;                 // 0..1023 16B chunks each
      stage16(klds + c * 8, Kg + c * 8);
      const int d = c >> 2, part = c & 3;
      stage16(vlds + d * 32 + part * 8, Vg + (size_t)d * TT + part * 8);
    }
    asm volatile("s_wait_asynccnt 0" ::: "memory");
    __syncthreads();

    if (kt < nkb) {                                // wave-uniform: EXEC stays ~0
      // ---- scores: S[16 x 32] = Q * K^T over D, two 16-col WMMA tiles -----
      v8f s0 = zero, s1 = zero;
      const __bf16* K0 = klds + ln * DD;
      const __bf16* K1 = klds + (16 + ln) * DD;
#pragma unroll
      for (int dc = 0; dc < 8; ++dc) {
        s0 = wmma_bf16(qf[dc], load_frag(K0 + dc * 32, kb), s0);
        s1 = wmma_bf16(qf[dc], load_frag(K1 + dc * 32, kb), s1);
      }

      // ---- online softmax (row's 16 columns live in one 16-lane half) -----
      const int kc0 = kt * 32 + ln, kc1 = kc0 + 16;
      float p0[8], p1[8];
#pragma unroll
      for (int rr = 0; rr < 8; ++rr) {
        const int qr = qrow0 + rr;
        float v0 = s0[rr] * sc, v1 = s1[rr] * sc;
        if (kc0 > qr) v0 = -1e30f;                 // causal mask
        if (kc1 > qr) v1 = -1e30f;
        float mx = fmaxf(v0, v1);
        mx = fmaxf(mx, __shfl_xor(mx, 1, 32));
        mx = fmaxf(mx, __shfl_xor(mx, 2, 32));
        mx = fmaxf(mx, __shfl_xor(mx, 4, 32));
        mx = fmaxf(mx, __shfl_xor(mx, 8, 32));
        const float mnew = fmaxf(mrow[rr], mx);
        const float corr = __builtin_exp2f((mrow[rr] - mnew) * LOG2E);
        mrow[rr] = mnew;
        p0[rr] = __builtin_exp2f((v0 - mnew) * LOG2E);
        p1[rr] = __builtin_exp2f((v1 - mnew) * LOG2E);
        float rs = p0[rr] + p1[rr];
        rs += __shfl_xor(rs, 1, 32);
        rs += __shfl_xor(rs, 2, 32);
        rs += __shfl_xor(rs, 4, 32);
        rs += __shfl_xor(rs, 8, 32);
        lrow[rr] = lrow[rr] * corr + rs;
#pragma unroll
        for (int nt = 0; nt < 16; ++nt) o[nt][rr] *= corr;
      }

      // ---- C-layout -> A-layout transpose of P through per-wave LDS -------
      __bf16* buf = plds[wslot];
#pragma unroll
      for (int rr = 0; rr < 8; ++rr) {
        const int row = rr + half * 8;
        buf[row * 32 + ln]      = (__bf16)p0[rr];
        buf[row * 32 + 16 + ln] = (__bf16)p1[rr];
      }
      asm volatile("s_wait_dscnt 0" ::: "memory");
      const v16bf pf = load_frag(buf + ln * 32, kb);

      // ---- O += P * V from the staged V^T panel ---------------------------
#pragma unroll
      for (int nt = 0; nt < 16; ++nt) {
        const v16bf vf = load_frag(vlds + (nt * 16 + ln) * 32, kb);
        o[nt] = wmma_bf16(pf, vf, o[nt]);
      }
    }
    __syncthreads();                               // protect LDS panel reuse
  }

  // ---- normalize and store heads concatenated: Zb [B][T][H*D] -------------
#pragma unroll
  for (int rr = 0; rr < 8; ++rr) {
    const float inv = 1.0f / lrow[rr];
    const int q = qt * 16 + rr + half * 8;
    __bf16* zr = Zb + ((size_t)b * TT + q) * EE + h * DD;
#pragma unroll
    for (int nt = 0; nt < 16; ++nt) zr[nt * 16 + ln] = (__bf16)(o[nt][rr] * inv);
  }
}

// ---------------------------------------------------------------------------
// Kernel 3: output projection  Out = Z @ Wo + bo (f32 result), same LDS
// weight-staging scheme as kernel 1: block = 8 row-tiles x one 64-col tile.
// ---------------------------------------------------------------------------
__global__ __launch_bounds__(256) void out_proj_kernel(
    const __bf16* __restrict__ Zb, const float* __restrict__ Wo,
    const float* __restrict__ bo, float* __restrict__ Out)
{
  __shared__ __align__(16) __bf16 wlds[64 * 32];

  const int tid  = threadIdx.x;
  const int lane = tid & 31;
  const int wave = tid >> 5;

  unsigned id = blockIdx.x;
  const int cg = id % (EE / 64);        // 12 col tiles of 64
  const int rg = id / (EE / 64);        // 64 row groups (8 tiles each)

  const int half = lane >> 4, ln = lane & 15, kb = half ? 8 : 0;

  const int rt = rg * 8 + wave;                           // 16-row tile
  const __bf16* Zrow = Zb + ((size_t)rt * 16 + ln) * EE;  // A rows

  const int dl = tid & 63;
  const int kq = tid >> 6;
  const float* Wcol = Wo + (size_t)cg * 64 + dl;

  v8f acc[4];
  v8f zero = {};
#pragma unroll
  for (int i = 0; i < 4; ++i) acc[i] = zero;

  for (int e0 = 0; e0 < EE; e0 += 32) {
    v8bf wv;
#pragma unroll
    for (int j = 0; j < 8; ++j)
      wv[j] = (__bf16)Wcol[(size_t)(e0 + kq * 8 + j) * EE];
    *(v8bf*)(wlds + dl * 32 + kq * 8) = wv;
    __syncthreads();

    const v16bf a = load_frag(Zrow + e0, kb);
#pragma unroll
    for (int i = 0; i < 4; ++i) {
      const v16bf bm = load_frag(wlds + (i * 16 + ln) * 32, kb);
      acc[i] = wmma_bf16(a, bm, acc[i]);
    }
    __syncthreads();
  }

#pragma unroll
  for (int i = 0; i < 4; ++i) {
    const int n = cg * 64 + i * 16 + ln;
    const float bias = bo[n];
#pragma unroll
    for (int rr = 0; rr < 8; ++rr) {
      const int m = rt * 16 + rr + half * 8;
      Out[(size_t)m * EE + n] = acc[i][rr] + bias;
    }
  }
}

// ---------------------------------------------------------------------------
extern "C" void kernel_launch(void* const* d_in, const int* in_sizes, int n_in,
                              void* d_out, int out_size, void* d_ws, size_t ws_size,
                              hipStream_t stream) {
  (void)in_sizes; (void)n_in; (void)out_size; (void)ws_size;
  const float* Xk = (const float*)d_in[0];
  const float* Xv = (const float*)d_in[1];
  const float* Xq = (const float*)d_in[2];
  const float* Wk = (const float*)d_in[3];
  const float* Wq = (const float*)d_in[4];
  const float* Wv = (const float*)d_in[5];
  const float* Wo = (const float*)d_in[6];
  const float* bo = (const float*)d_in[7];
  float* Out = (float*)d_out;

  char* ws = (char*)d_ws;
  const size_t szQ = (size_t)BB * HH * TT * DD * sizeof(__bf16);  // 12.6 MB
  __bf16* Qb = (__bf16*)(ws);
  __bf16* Kb = (__bf16*)(ws + szQ);
  __bf16* Vt = (__bf16*)(ws + 2 * szQ);
  __bf16* Zb = (__bf16*)(ws + 3 * szQ);                 // [B][T][E] bf16

  // 3 types * H * (D/64) * B * 16 t-groups = 2304 blocks, 256 threads
  qkv_proj_kernel<<<3 * HH * (DD / 64) * BB * (TT / 16 / 8), 256, 0, stream>>>(
      Xk, Xv, Xq, Wk, Wq, Wv, Qb, Kb, Vt);
  // B*H*(T/16) = 1536 q-tiles, 4 waves / 128-thread block
  attn_kernel<<<(BB * HH * (TT / 16)) / 4, 128, 0, stream>>>(Qb, Kb, Vt, Zb);
  // (E/64) * (B*T/16/8) = 12 * 64 = 768 blocks, 256 threads
  out_proj_kernel<<<(EE / 64) * (BB * TT / 16 / 8), 256, 0, stream>>>(
      Zb, Wo, bo, Out);
}
